// KernelNN_39462159516209
// MI455X (gfx1250) — matrix operand order repaired and checked by reading, not compiled, not run on hardware
//
// MI455X (gfx1250) implementation of a multipole GNN forward pass.
//
// Design: the per-edge kernel-MLP hidden state H[e,64] is loop-invariant and
// computed once with WMMA GEMMs (f16 16x16x32). The per-edge message einsum
//   msg[e,o] = sum_{k,i} H[e,k]*x[src,i]*W3[k,i,o]
// is a GEMM with a SHARED B [65*32, out] whose A rows are rank-1 products
// generated in-register: per K-step the A fragment is the gathered x_src
// fragment scaled by the per-lane scalar H[e,kk] (8x v_pk_mul_f16) feeding
// v_wmma_f32_16x16x32_f16. This avoids materializing the 512MB of per-edge
// 32x32 matrices. Scatter-mean uses global f32 atomics into L2-resident node
// sums. in=2 convs use precomputed per-edge weights (WMMA GEMM) + VALU
// messages. The inner-conv x@root+bias term is also a WMMA GEMM over the f16
// node mirror. Requires ~105 MB of workspace.

#include <hip/hip_runtime.h>
#include <stdint.h>

#define N1 4096
#define N2 256
#define N3 4096
#define NTOT (N1 + N2 + N3)
#define E1c 131072
#define E2c 8192
#define E3c 131072
#define EXc 32768

typedef __attribute__((ext_vector_type(16))) _Float16 v16h;
typedef __attribute__((ext_vector_type(8)))  float    v8f;
typedef __attribute__((ext_vector_type(2)))  _Float16 h2;

union ABf { uint4 q[2]; unsigned u[8]; v16h v; };

__device__ __forceinline__ unsigned short f2h(float f) {
  _Float16 h = (_Float16)f;  // v_cvt_f16_f32 (RNE)
  return __builtin_bit_cast(unsigned short, h);
}

__device__ __forceinline__ unsigned pkmul(unsigned a, unsigned b) {
  h2 x = __builtin_bit_cast(h2, a);
  h2 y = __builtin_bit_cast(h2, b);
  h2 r = x * y;  // v_pk_mul_f16
  return __builtin_bit_cast(unsigned, r);
}

// ---------------- utility kernels ----------------

__global__ void k_zero_f(float* p, int n) {
  int t = blockIdx.x * blockDim.x + threadIdx.x;
  if (t < n) p[t] = 0.f;
}
__global__ void k_zero_i(int* p, int n) {
  int t = blockIdx.x * blockDim.x + threadIdx.x;
  if (t < n) p[t] = 0;
}
__global__ void k_count(const int* __restrict__ dst, int E, int lo, int hi,
                        int* __restrict__ cnt) {
  int t = blockIdx.x * blockDim.x + threadIdx.x;
  if (t >= E) return;
  int d = dst[t];
  if (d >= lo && d < hi) atomicAdd(&cnt[d - lo], 1);
}
__global__ void k_invc(const int* __restrict__ cnt, float* __restrict__ invc, int n) {
  int t = blockIdx.x * blockDim.x + threadIdx.x;
  if (t >= n) return;
  int c = cnt[t];
  invc[t] = 1.0f / (float)(c > 0 ? c : 1);
}

// edge attrs f32[E,6] -> f16[E,32]: cols 0..5 = attr, col6 = 1.0 (bias hook), rest 0
__global__ void k_attr2h(const float* __restrict__ ea, unsigned short* __restrict__ A, int E) {
  long long t = (long long)blockIdx.x * blockDim.x + threadIdx.x;
  if (t >= (long long)E * 32) return;
  int e = (int)(t >> 5), c = (int)(t & 31);
  float v = (c < 6) ? ea[(size_t)e * 6 + c] : (c == 6 ? 1.f : 0.f);
  A[(size_t)e * 32 + c] = f2h(v);
}

// H fixup: col 64 = 1.0 (bias row hook for W3), cols 65..pitch-1 = 0
__global__ void k_fixH(unsigned short* H, int E, int pitch) {
  int ext = pitch - 64;
  long long t = (long long)blockIdx.x * blockDim.x + threadIdx.x;
  if (t >= (long long)E * ext) return;
  int e = (int)(t / ext);
  int c = 64 + (int)(t - (long long)e * ext);
  H[(size_t)e * pitch + c] = (c == 64) ? (unsigned short)0x3C00 : (unsigned short)0;
}

// Pack a row-major f32 matrix W[realK, realN] (optional bias row at K==realK)
// into WMMA B-fragment blocks: per (kstep,tile) a 1KB block; lane L holds 16
// f16: slot j <-> (K_local = (L<16?0:8)+(j&7)+(j&8?16:0), N = tile*16 + (L&15)).
__global__ void k_pack_std(unsigned short* __restrict__ dst, const float* __restrict__ W,
                           const float* __restrict__ biasRow,
                           int realK, int realN, int ksteps, int ntiles) {
  long long t = (long long)blockIdx.x * blockDim.x + threadIdx.x;
  long long tot = (long long)ksteps * ntiles * 512;
  if (t >= tot) return;
  int blk = (int)(t >> 9), s = (int)(t & 511);
  int kk = blk / ntiles, tile = blk - kk * ntiles;
  int lane = s >> 4, j = s & 15;
  int K = kk * 32 + ((lane & 16) ? 8 : 0) + (j & 7) + ((j & 8) ? 16 : 0);
  int N = tile * 16 + (lane & 15);
  float v = 0.f;
  if (N < realN) {
    if (K < realK) v = W[(size_t)K * realN + N];
    else if (K == realK && biasRow) v = biasRow[N];
  }
  dst[(size_t)blk * 512 + s] = f2h(v);
}

// Pack W3[64, 32*out] (+b3 as K-step 64) for the fused message GEMM:
// global K = kk*32 + i, B[K, o] = (kk<64 ? W3[kk, i*out+o] : b3[i*out+o]).
__global__ void k_pack_w3f(unsigned short* __restrict__ dst, const float* __restrict__ W3,
                           const float* __restrict__ b3, int outc, int ntiles) {
  long long t = (long long)blockIdx.x * blockDim.x + threadIdx.x;
  long long tot = (long long)65 * ntiles * 512;
  if (t >= tot) return;
  int blk = (int)(t >> 9), s = (int)(t & 511);
  int kk = blk / ntiles, tile = blk - kk * ntiles;
  int lane = s >> 4, j = s & 15;
  int i = ((lane & 16) ? 8 : 0) + (j & 7) + ((j & 8) ? 16 : 0);
  int o = tile * 16 + (lane & 15);
  float v = 0.f;
  if (o < outc)
    v = (kk < 64) ? W3[(size_t)kk * 32 * outc + (size_t)i * outc + o]
                  : b3[(size_t)i * outc + o];
  dst[(size_t)blk * 512 + s] = f2h(v);
}

// ---------------- WMMA GEMM: C[M, realN] = op(A[M, K] @ Bpacked + bias) ----------------
// A f16 row-major (lda in elements, rows multiple of 16), B pre-packed.
// flags: 1 = relu, 2 = f16 output (else f32).
__global__ void __launch_bounds__(256)
k_gemm(const unsigned short* __restrict__ A, int lda,
       const unsigned short* __restrict__ Bp,
       const float* __restrict__ bias,
       void* __restrict__ C, int ldc, int realN,
       int M, int ksteps, int ntiles, int flags) {
  int lane = threadIdx.x & 31;
  int wid  = blockIdx.x * (blockDim.x >> 5) + (threadIdx.x >> 5);
  if (wid >= (M >> 4) * ntiles) return;
  int gidx = wid / ntiles, tile = wid - gidx * ntiles;
  int row  = gidx * 16 + (lane & 15);
  int offk = (lane & 16) ? 8 : 0;
  const unsigned short* Ar = A + (size_t)row * lda + offk;
  const uint4* bb = (const uint4*)Bp + (size_t)tile * 64 + lane * 2;
  v8f acc = {};
  for (int kk = 0; kk < ksteps; ++kk) {
    ABf a, b;
    a.q[0] = *(const uint4*)(Ar + kk * 32);
    a.q[1] = *(const uint4*)(Ar + kk * 32 + 16);
    const uint4* bp = bb + (size_t)kk * ntiles * 64;
    b.q[0] = bp[0];
    b.q[1] = bp[1];
    acc = __builtin_amdgcn_wmma_f32_16x16x32_f16(false, a.v, false, b.v,
                                                 (short)0, acc, false, false);
  }
  int col = tile * 16 + (lane & 15);
  if (col >= realN) return;
  float bv = bias ? bias[col] : 0.f;
  int r0 = gidx * 16 + ((lane >> 4) << 3);
  #pragma unroll
  for (int r = 0; r < 8; ++r) {
    float v = acc[r] + bv;
    if (flags & 1) v = fmaxf(v, 0.f);
    if (flags & 2) ((unsigned short*)C)[(size_t)(r0 + r) * ldc + col] = f2h(v);
    else           ((float*)C)[(size_t)(r0 + r) * ldc + col] = v;
  }
}

// ---------------- fused message GEMM for in=32 convs ----------------
// One wave per 16-edge group. Per K-step kk the A fragment is the gathered
// x_src fragment scaled by H[e,kk] (v_pk_mul_f16 x8), then WMMA against the
// shared packed B. Accumulated D is scattered with f32 atomics (segment sum).
// Gather rule (concat [gA | zeros(nb) | gB]): g<na -> gA, g<na+nb -> 0, else gB.
template <int NTILES>
__global__ void __launch_bounds__(256)
k_msg32(const int* __restrict__ src, const int* __restrict__ dst, int E,
        const unsigned short* __restrict__ H,   // pitch 68, col64 = 1.0
        const unsigned short* __restrict__ Bp,  // 65*NTILES blocks of 1KB
        const unsigned short* __restrict__ gA, int na, int nb,
        const unsigned short* __restrict__ gB,
        int outc, int dstLo, int dstHi, float* __restrict__ S) {
  __shared__ unsigned short sh[8][16 * 68];
  int lane = threadIdx.x & 31;
  int wv   = threadIdx.x >> 5;
  int eg   = (blockIdx.x * 8 + wv) * 16;
  bool act = eg < E;
  if (act) {  // stage 16 H rows (16*68 f16 contiguous) into LDS
    const unsigned* hs = (const unsigned*)(H + (size_t)eg * 68);
    unsigned* sw = (unsigned*)&sh[wv][0];
    for (int t = lane; t < 544; t += 32) sw[t] = hs[t];
  }
  __syncthreads();
  if (!act) return;

  int m = lane & 15;
  int g = src[eg + m];
  const unsigned short* xr = 0;
  if (g < na) xr = gA + (size_t)g * 32;
  else if (g >= na + nb) xr = gB + (size_t)(g - na - nb) * 32;

  ABf xs;
  int offk = (lane & 16) ? 8 : 0;
  if (xr) {
    xs.q[0] = *(const uint4*)(xr + offk);
    xs.q[1] = *(const uint4*)(xr + offk + 16);
  } else {
    xs.q[0] = make_uint4(0u, 0u, 0u, 0u);
    xs.q[1] = make_uint4(0u, 0u, 0u, 0u);
  }

  v8f c0 = {};
  v8f c1 = {};
  const unsigned short* hr = &sh[wv][m * 68];
  const uint4* bb = (const uint4*)Bp + lane * 2;
  for (int kk = 0; kk < 65; ++kk) {
    unsigned h = hr[kk];
    unsigned hh = h | (h << 16);
    ABf a;
    #pragma unroll
    for (int j = 0; j < 8; ++j) a.u[j] = pkmul(xs.u[j], hh);
    {
      ABf b;
      const uint4* bp = bb + (size_t)(kk * NTILES) * 64;
      b.q[0] = bp[0]; b.q[1] = bp[1];
      c0 = __builtin_amdgcn_wmma_f32_16x16x32_f16(false, a.v, false, b.v,
                                                  (short)0, c0, false, false);
    }
    if (NTILES == 2) {
      ABf b;
      const uint4* bp = bb + (size_t)(kk * NTILES + 1) * 64;
      b.q[0] = bp[0]; b.q[1] = bp[1];
      c1 = __builtin_amdgcn_wmma_f32_16x16x32_f16(false, a.v, false, b.v,
                                                  (short)0, c1, false, false);
    }
  }

  // cache the 8 destination indices owned by this lane once
  int mb = (lane >> 4) << 3;
  int dn[8];
  #pragma unroll
  for (int r = 0; r < 8; ++r) dn[r] = dst[eg + mb + r];

  int col = lane & 15;
  if (col < outc) {
    #pragma unroll
    for (int r = 0; r < 8; ++r)
      if (dn[r] >= dstLo && dn[r] < dstHi)
        atomicAdd(S + (size_t)(dn[r] - dstLo) * outc + col, c0[r]);
  }
  if (NTILES == 2 && (16 + col) < outc) {
    #pragma unroll
    for (int r = 0; r < 8; ++r)
      if (dn[r] >= dstLo && dn[r] < dstHi)
        atomicAdd(S + (size_t)(dn[r] - dstLo) * outc + 16 + col, c1[r]);
  }
}

// in=2 messages with precomputed per-edge weights we[E, 2*out] (i-major).
__global__ void k_msg2(const int* __restrict__ src, const int* __restrict__ dst, int E,
                       const float* __restrict__ we, int ld, int oshift,
                       int glo, int ghi, const float* __restrict__ x2v,
                       int dstLo, int dstHi, float* __restrict__ S, int outc) {
  long long t = (long long)blockIdx.x * blockDim.x + threadIdx.x;
  if (t >= ((long long)E << oshift)) return;
  int e = (int)(t >> oshift), o = (int)(t & (outc - 1));
  int g = src[e];
  float s0 = 0.f, s1 = 0.f;
  if (g >= glo && g < ghi) {
    s0 = x2v[(size_t)(g - glo) * 2];
    s1 = x2v[(size_t)(g - glo) * 2 + 1];
  }
  float msg = s0 * we[(size_t)e * ld + o] + s1 * we[(size_t)e * ld + outc + o];
  int d = dst[e];
  if (d >= dstLo && d < dstHi)
    atomicAdd(S + (size_t)(d - dstLo) * outc + o, msg);
}

// finishes ---------------------------------------------------------------

// inner width-32 conv: x = relu(S*invc + R), R = x_old@root + bias (WMMA GEMM).
__global__ void k_finish32(const float* __restrict__ S, const float* __restrict__ invc,
                           const float* __restrict__ R, int n,
                           float* __restrict__ x, unsigned short* __restrict__ xb) {
  long long t = (long long)blockIdx.x * blockDim.x + threadIdx.x;
  if (t >= (long long)n * 32) return;
  int node = (int)(t >> 5);
  float v = fmaxf(S[t] * invc[node] + R[t], 0.f);
  x[t] = v;
  xb[t] = f2h(v);
}

// inner width-2 conv (conv2)
__global__ void k_finish2(const float* __restrict__ S, const float* __restrict__ invc,
                          const float* __restrict__ root, const float* __restrict__ bias,
                          int n, float* __restrict__ x) {
  int node = blockIdx.x * blockDim.x + threadIdx.x;
  if (node >= n) return;
  float a = x[node * 2], b = x[node * 2 + 1], ic = invc[node];
  float v0 = fmaxf(S[node * 2] * ic + bias[0] + a * root[0] + b * root[2], 0.f);
  float v1 = fmaxf(S[node * 2 + 1] * ic + bias[1] + a * root[1] + b * root[3], 0.f);
  x[node * 2] = v0;
  x[node * 2 + 1] = v1;
}

// inter conv accumulate: x += relu(S*invc + bias)   (root term vanishes: zero slice)
__global__ void k_finishAcc(const float* __restrict__ S, const float* __restrict__ invc,
                            const float* __restrict__ bias, int n, int outc,
                            float* __restrict__ x, unsigned short* __restrict__ xb) {
  long long t = (long long)blockIdx.x * blockDim.x + threadIdx.x;
  if (t >= (long long)n * outc) return;
  int node = (int)(t / outc), o = (int)(t - (long long)node * outc);
  float v = x[t] + fmaxf(S[t] * invc[node] + bias[o], 0.f);
  x[t] = v;
  if (xb) xb[t] = f2h(v);
}

__global__ void k_fc1(const float* __restrict__ x, const float* __restrict__ W,
                      const float* __restrict__ b, int n,
                      float* __restrict__ x1, unsigned short* __restrict__ x1b) {
  long long t = (long long)blockIdx.x * blockDim.x + threadIdx.x;
  if (t >= (long long)n * 32) return;
  int node = (int)(t >> 5), o = (int)(t & 31);
  float v = x[node] * W[o] + b[o];
  x1[t] = v;
  x1b[t] = f2h(v);
}

__global__ void k_out(const float* __restrict__ x3, const float* __restrict__ W,
                      const float* __restrict__ b, int n, float* __restrict__ out) {
  int node = blockIdx.x * blockDim.x + threadIdx.x;
  if (node >= n) return;
  float s = b[0];
  #pragma unroll
  for (int i = 0; i < 32; ++i) s += x3[(size_t)node * 32 + i] * W[i];
  out[node] = s;
}

// ---------------- host orchestration ----------------

extern "C" void kernel_launch(void* const* d_in, const int* in_sizes, int n_in,
                              void* d_out, int out_size, void* d_ws, size_t ws_size,
                              hipStream_t stream) {
  (void)in_sizes; (void)n_in; (void)out_size; (void)ws_size;
  const float* xin = (const float*)d_in[0];
  const int* ei[7];
  const float* ea[7];
  for (int i = 0; i < 7; ++i) {
    ei[i] = (const int*)d_in[1 + i];
    ea[i] = (const float*)d_in[8 + i];
  }
  static const int Eset[7] = {E1c, E2c, E3c, EXc, EXc, EXc, EXc};
  // params flattened in sorted-pytree order: conv1, conv12, conv2, conv21,
  // conv23, conv3, conv32; each: bias, l1.W, l1.b, l2.W, l2.b, l3.W, l3.b, root
  const float* P[7][8];
  for (int c = 0; c < 7; ++c)
    for (int l = 0; l < 8; ++l) P[c][l] = (const float*)d_in[15 + c * 8 + l];
  const float* fc1W = (const float*)d_in[71];
  const float* fc1b = (const float*)d_in[72];
  const float* fc2W = (const float*)d_in[73];
  const float* fc2b = (const float*)d_in[74];

  static const int cset[7]  = {0, 3, 1, 4, 5, 2, 6};         // edge set per conv
  static const int cin_[7]  = {32, 32, 2, 2, 2, 32, 32};
  static const int cout_[7] = {32, 2, 2, 32, 32, 32, 2};
  static const int clo_[7]  = {0, N1, 0, 0, N1 + N2, 0, N1}; // dst slice
  static const int chi_[7]  = {N1, N1 + N2, N2, N1, NTOT, N3, N1 + N2};

  char* wsb = (char*)d_ws;
  size_t off = 0;
  auto alloc = [&](size_t bytes) -> void* {
    off = (off + 255) & ~(size_t)255;
    void* p = wsb + off;
    off += bytes;
    return p;
  };

  float* x1 = (float*)alloc((size_t)N1 * 32 * 4);
  float* x3 = (float*)alloc((size_t)N3 * 32 * 4);
  float* x2 = (float*)alloc((size_t)N2 * 2 * 4);
  float* S  = (float*)alloc((size_t)N3 * 32 * 4);
  float* Rb = (float*)alloc((size_t)N3 * 32 * 4);   // root-term GEMM output
  unsigned short* x1b = (unsigned short*)alloc((size_t)N1 * 32 * 2);
  unsigned short* x3b = (unsigned short*)alloc((size_t)N3 * 32 * 2);
  int* cnt = (int*)alloc(4096 * 4);
  unsigned short* Abuf = (unsigned short*)alloc((size_t)E1c * 32 * 2);
  unsigned short* H1t  = (unsigned short*)alloc((size_t)E1c * 64 * 2);

  unsigned short* Hbuf[7];
  unsigned short* Bp[7];
  unsigned short *W1p[7], *W2p[7], *W3p[7], *RootP[7];
  float* weB[7];
  float* invc[7];
  for (int c = 0; c < 7; ++c) {
    int E = Eset[cset[c]];
    int pitch = (cin_[c] == 32) ? 68 : 96;
    Hbuf[c] = (unsigned short*)alloc((size_t)E * pitch * 2);
    invc[c] = (float*)alloc((size_t)(chi_[c] - clo_[c]) * 4);
    W1p[c] = (unsigned short*)alloc(4 * 1024);
    W2p[c] = (unsigned short*)alloc(8 * 1024);
    RootP[c] = nullptr;
    if (cin_[c] == 32) {
      int nt = (cout_[c] + 15) / 16;
      Bp[c] = (unsigned short*)alloc((size_t)65 * nt * 1024);
      weB[c] = nullptr;
      W3p[c] = nullptr;
      if (cout_[c] == 32) RootP[c] = (unsigned short*)alloc(2 * 1024);
    } else {
      int n2o = 2 * cout_[c];
      int nt = (n2o + 15) / 16;
      W3p[c] = (unsigned short*)alloc((size_t)3 * nt * 1024);
      weB[c] = (float*)alloc((size_t)E * n2o * 4);
      Bp[c] = nullptr;
    }
  }

  auto gr = [](long long n) -> dim3 { return dim3((unsigned)((n + 255) / 256)); };

  // ---- one-time (per launch) precompute: counts, H, packed B / per-edge we ----
  for (int c = 0; c < 7; ++c) {
    int s = cset[c], E = Eset[s];
    int pitch = (cin_[c] == 32) ? 68 : 96;
    const int* dstI = ei[s] + E;
    int sn = chi_[c] - clo_[c];
    k_zero_i<<<gr(sn), 256, 0, stream>>>(cnt, sn);
    k_count<<<gr(E), 256, 0, stream>>>(dstI, E, clo_[c], chi_[c], cnt);
    k_invc<<<gr(sn), 256, 0, stream>>>(cnt, invc[c], sn);

    k_attr2h<<<gr((long long)E * 32), 256, 0, stream>>>(ea[s], Abuf, E);
    k_pack_std<<<gr(4 * 512), 256, 0, stream>>>(W1p[c], P[c][1], P[c][2], 6, 64, 1, 4);
    k_gemm<<<gr((long long)(E / 16) * 4 * 32), 256, 0, stream>>>(
        Abuf, 32, W1p[c], nullptr, H1t, 64, 64, E, 1, 4, 1 | 2);
    k_pack_std<<<gr(8 * 512), 256, 0, stream>>>(W2p[c], P[c][3], nullptr, 64, 64, 2, 4);
    k_gemm<<<gr((long long)(E / 16) * 4 * 32), 256, 0, stream>>>(
        H1t, 64, W2p[c], P[c][4], Hbuf[c], pitch, 64, E, 2, 4, 1 | 2);
    k_fixH<<<gr((long long)E * (pitch - 64)), 256, 0, stream>>>(Hbuf[c], E, pitch);

    if (cin_[c] == 32) {
      int nt = (cout_[c] + 15) / 16;
      k_pack_w3f<<<gr((long long)65 * nt * 512), 256, 0, stream>>>(
          Bp[c], P[c][5], P[c][6], cout_[c], nt);
      if (RootP[c])
        k_pack_std<<<gr(2 * 512), 256, 0, stream>>>(RootP[c], P[c][7], nullptr, 32, 32, 1, 2);
    } else {
      int n2o = 2 * cout_[c];
      int nt = (n2o + 15) / 16;
      k_pack_std<<<gr((long long)3 * nt * 512), 256, 0, stream>>>(
          W3p[c], P[c][5], P[c][6], 64, n2o, 3, nt);
      k_gemm<<<gr((long long)(E / 16) * nt * 32), 256, 0, stream>>>(
          Hbuf[c], 96, W3p[c], nullptr, weB[c], n2o, n2o, E, 3, nt, 0);
    }
  }

  // ---- init node states ----
  k_fc1<<<gr((long long)N1 * 32), 256, 0, stream>>>(xin, fc1W, fc1b, N1, x1, x1b);
  k_zero_f<<<gr(N2 * 2), 256, 0, stream>>>(x2, N2 * 2);
  k_zero_f<<<gr(N3 * 32), 256, 0, stream>>>(x3, N3 * 32);
  k_zero_f<<<gr(N3 * 16), 256, 0, stream>>>((float*)x3b, N3 * 16);

  // ---- DEPTH iterations ----
  for (int it = 0; it < 3; ++it) {
    // A: conv1 (inner1, 32->32)
    k_gemm<<<gr((long long)(N1 / 16) * 2 * 32), 256, 0, stream>>>(
        x1b, 32, RootP[0], P[0][0], Rb, 32, 32, N1, 1, 2, 0);  // R = x1@root + bias
    k_zero_f<<<gr(N1 * 32), 256, 0, stream>>>(S, N1 * 32);
    k_msg32<2><<<dim3(E1c / 128), 256, 0, stream>>>(
        ei[0], ei[0] + E1c, E1c, Hbuf[0], Bp[0], x1b, N1, 0, x1b, 32, 0, N1, S);
    k_finish32<<<gr((long long)N1 * 32), 256, 0, stream>>>(S, invc[0], Rb, N1, x1, x1b);

    // B: conv12 (inter12, wide gather [x1|0|x3], 32->2), x2 += relu(...)
    k_zero_f<<<gr(N2 * 2), 256, 0, stream>>>(S, N2 * 2);
    k_msg32<1><<<dim3(EXc / 128), 256, 0, stream>>>(
        ei[3], ei[3] + EXc, EXc, Hbuf[1], Bp[1], x1b, N1, N2, x3b, 2, N1, N1 + N2, S);
    k_finishAcc<<<gr(N2 * 2), 256, 0, stream>>>(S, invc[1], P[1][0], N2, 2, x2,
                                                (unsigned short*)nullptr);

    // C: conv2 (inner2, 2->2)
    k_zero_f<<<gr(N2 * 2), 256, 0, stream>>>(S, N2 * 2);
    k_msg2<<<gr((long long)E2c * 2), 256, 0, stream>>>(
        ei[1], ei[1] + E2c, E2c, weB[2], 4, 1, 0, N2, x2, 0, N2, S, 2);
    k_finish2<<<gr(N2), 256, 0, stream>>>(S, invc[2], P[2][7], P[2][0], N2, x2);

    // D: conv23 (inter23, narrow gather x2, 2->32), x3 += relu(...)
    k_zero_f<<<gr(N3 * 32), 256, 0, stream>>>(S, N3 * 32);
    k_msg2<<<gr((long long)EXc * 32), 256, 0, stream>>>(
        ei[5], ei[5] + EXc, EXc, weB[4], 64, 5, N1, N1 + N2, x2, N1 + N2, NTOT, S, 32);
    k_finishAcc<<<gr((long long)N3 * 32), 256, 0, stream>>>(S, invc[4], P[4][0], N3, 32,
                                                            x3, x3b);

    // E: conv3 (inner3, 32->32)
    k_gemm<<<gr((long long)(N3 / 16) * 2 * 32), 256, 0, stream>>>(
        x3b, 32, RootP[5], P[5][0], Rb, 32, 32, N3, 1, 2, 0);  // R = x3@root + bias
    k_zero_f<<<gr(N3 * 32), 256, 0, stream>>>(S, N3 * 32);
    k_msg32<2><<<dim3(E3c / 128), 256, 0, stream>>>(
        ei[2], ei[2] + E3c, E3c, Hbuf[5], Bp[5], x3b, N3, 0, x3b, 32, 0, N3, S);
    k_finish32<<<gr((long long)N3 * 32), 256, 0, stream>>>(S, invc[5], Rb, N3, x3, x3b);

    // F: conv32 (inter32, wide gather, 32->2), x2 += relu(...)
    k_zero_f<<<gr(N2 * 2), 256, 0, stream>>>(S, N2 * 2);
    k_msg32<1><<<dim3(EXc / 128), 256, 0, stream>>>(
        ei[6], ei[6] + EXc, EXc, Hbuf[6], Bp[6], x1b, N1, N2, x3b, 2, N1, N1 + N2, S);
    k_finishAcc<<<gr(N2 * 2), 256, 0, stream>>>(S, invc[6], P[6][0], N2, 2, x2,
                                                (unsigned short*)nullptr);

    // G: conv2 again
    k_zero_f<<<gr(N2 * 2), 256, 0, stream>>>(S, N2 * 2);
    k_msg2<<<gr((long long)E2c * 2), 256, 0, stream>>>(
        ei[1], ei[1] + E2c, E2c, weB[2], 4, 1, 0, N2, x2, 0, N2, S, 2);
    k_finish2<<<gr(N2), 256, 0, stream>>>(S, invc[2], P[2][7], P[2][0], N2, x2);

    // H: conv21 (inter21, narrow gather x2, 2->32), x1 += relu(...)
    k_zero_f<<<gr(N1 * 32), 256, 0, stream>>>(S, N1 * 32);
    k_msg2<<<gr((long long)EXc * 32), 256, 0, stream>>>(
        ei[4], ei[4] + EXc, EXc, weB[3], 64, 5, N1, N1 + N2, x2, 0, N1, S, 32);
    k_finishAcc<<<gr((long long)N1 * 32), 256, 0, stream>>>(S, invc[3], P[3][0], N1, 32,
                                                            x1, x1b);
  }

  // output head: out = x3 @ fc2.W + fc2.b
  k_out<<<gr(N3), 256, 0, stream>>>(x3, fc2W, fc2b, N3, (float*)d_out);
}
